// RNN_635655160000
// MI455X (gfx1250) — compile-verified, matrix operand order checked by reading, not run on hardware
//
#include <hip/hip_runtime.h>
#include <hip/hip_bf16.h>
#include <math.h>

typedef __attribute__((ext_vector_type(16))) _Float16 v16h;
typedef __attribute__((ext_vector_type(8)))  _Float16 v8h;
typedef __attribute__((ext_vector_type(8)))  float    v8f;

#define RNN_B 256
#define RNN_T 128
#define RNN_I 64
#define RNN_H 1024
#define RNN_C 64
#define RNN_F 32

// ---------------- helper kernels ----------------

__global__ void f32_to_f16_kernel(const float* __restrict__ src,
                                  _Float16* __restrict__ dst, int n) {
  int i = blockIdx.x * blockDim.x + threadIdx.x;
  if (i < n) dst[i] = (_Float16)src[i];
}

__global__ void bias_combine_kernel(const float* __restrict__ a,
                                    const float* __restrict__ b,
                                    float* __restrict__ out, int n) {
  int i = blockIdx.x * blockDim.x + threadIdx.x;
  if (i < n) out[i] = a[i] + b[i];
}

__global__ void zero_f16_kernel(_Float16* __restrict__ p, int n) {
  int i = blockIdx.x * blockDim.x + threadIdx.x;
  if (i < n) p[i] = (_Float16)0.0f;
}

// ---------------- WMMA fragment load ----------------
// 16-bit A 16x32 layout (wave32):
//   lanes 0-15  : M = lane,    halves 0-7 -> K=0..7,  halves 8-15 -> K=16..23
//   lanes 16-31 : M = lane-16, halves 0-7 -> K=8..15, halves 8-15 -> K=24..31
// => per lane: two contiguous 8-half (16B) loads at k0 and k0+16, with
//    k0 = kbase + (lane>=16 ? 8 : 0).  B 32x16 mirrors this with N = lane%16
//    and B[k][n] = W[n][k] contiguous along k (row-major W).
__device__ __forceinline__ v16h load_frag16(const _Float16* __restrict__ p) {
  v8h lo = *(const v8h*)(p);
  v8h hi = *(const v8h*)(p + 16);
  v16h r;
#pragma unroll
  for (int i = 0; i < 8; ++i) { r[i] = lo[i]; r[i + 8] = hi[i]; }
  return r;
}

// ---------------- fused RNN layer: Hnext = tanh(X*Wih^T + Hprev*Whh^T + bias) ----------------
// 2x2 register-blocked: each wave computes a 32x32 output patch (4 WMMA tiles)
// from 2 A-fragments and 2 B-fragments per K-step => 2x arithmetic intensity.
// X: (B, Kx) f16 with row stride ldx; Wih: (H, Kx) f16 row-major
// Hprev: (B, H) f16; Whh: (H, H) f16; bias: (H) f32; Hnext: (B, H) f16
__global__ void __launch_bounds__(256)
rnn_layer_kernel(const _Float16* __restrict__ X, int ldx, int Kx,
                 const _Float16* __restrict__ Wih,
                 const _Float16* __restrict__ Hprev,
                 const _Float16* __restrict__ Whh,
                 const float* __restrict__ bias,
                 _Float16* __restrict__ Hnext) {
  const int NPAIR = RNN_H / 32;  // 32 column-pairs of 16 => 32 N-groups
  int wave = blockIdx.x * (blockDim.x >> 5) + (threadIdx.x >> 5);
  int tm   = wave / NPAIR;       // 0..7  (row group of 32)
  int tn   = wave % NPAIR;       // 0..31 (col group of 32)
  int lane = threadIdx.x & 31;
  int l15  = lane & 15;
  int ksel = (lane >> 4) << 3;   // 0 or 8

  int mrow0 = tm * 32 + l15;
  int mrow1 = mrow0 + 16;
  int ncol0 = tn * 32 + l15;
  int ncol1 = ncol0 + 16;

  v8f acc00 = {}, acc01 = {}, acc10 = {}, acc11 = {};

  // X * Wih^T contribution
  {
    const _Float16* a0 = X   + (size_t)mrow0 * ldx + ksel;
    const _Float16* a1 = X   + (size_t)mrow1 * ldx + ksel;
    const _Float16* b0 = Wih + (size_t)ncol0 * Kx  + ksel;
    const _Float16* b1 = Wih + (size_t)ncol1 * Kx  + ksel;
    for (int k = 0; k < Kx; k += 32) {
      v16h fa0 = load_frag16(a0 + k);
      v16h fa1 = load_frag16(a1 + k);
      v16h fb0 = load_frag16(b0 + k);
      v16h fb1 = load_frag16(b1 + k);
      acc00 = __builtin_amdgcn_wmma_f32_16x16x32_f16(false, fa0, false, fb0, (short)0, acc00, false, false);
      acc01 = __builtin_amdgcn_wmma_f32_16x16x32_f16(false, fa0, false, fb1, (short)0, acc01, false, false);
      acc10 = __builtin_amdgcn_wmma_f32_16x16x32_f16(false, fa1, false, fb0, (short)0, acc10, false, false);
      acc11 = __builtin_amdgcn_wmma_f32_16x16x32_f16(false, fa1, false, fb1, (short)0, acc11, false, false);
    }
  }
  // Hprev * Whh^T contribution
  {
    const _Float16* a0 = Hprev + (size_t)mrow0 * RNN_H + ksel;
    const _Float16* a1 = Hprev + (size_t)mrow1 * RNN_H + ksel;
    const _Float16* b0 = Whh   + (size_t)ncol0 * RNN_H + ksel;
    const _Float16* b1 = Whh   + (size_t)ncol1 * RNN_H + ksel;
    for (int k = 0; k < RNN_H; k += 32) {
      v16h fa0 = load_frag16(a0 + k);
      v16h fa1 = load_frag16(a1 + k);
      v16h fb0 = load_frag16(b0 + k);
      v16h fb1 = load_frag16(b1 + k);
      acc00 = __builtin_amdgcn_wmma_f32_16x16x32_f16(false, fa0, false, fb0, (short)0, acc00, false, false);
      acc01 = __builtin_amdgcn_wmma_f32_16x16x32_f16(false, fa0, false, fb1, (short)0, acc01, false, false);
      acc10 = __builtin_amdgcn_wmma_f32_16x16x32_f16(false, fa1, false, fb0, (short)0, acc10, false, false);
      acc11 = __builtin_amdgcn_wmma_f32_16x16x32_f16(false, fa1, false, fb1, (short)0, acc11, false, false);
    }
  }

  // Epilogue: C layout => VGPR r holds M = r (lanes 0-15) / r+8 (lanes 16-31), N = lane%16
  float bn0   = bias[ncol0];
  float bn1   = bias[ncol1];
  int   mb0   = tm * 32 + ((lane < 16) ? 0 : 8);
  int   mb1   = mb0 + 16;
#pragma unroll
  for (int r = 0; r < 8; ++r) {
    Hnext[(size_t)(mb0 + r) * RNN_H + ncol0] = (_Float16)tanhf(acc00[r] + bn0);
    Hnext[(size_t)(mb0 + r) * RNN_H + ncol1] = (_Float16)tanhf(acc01[r] + bn1);
    Hnext[(size_t)(mb1 + r) * RNN_H + ncol0] = (_Float16)tanhf(acc10[r] + bn0);
    Hnext[(size_t)(mb1 + r) * RNN_H + ncol1] = (_Float16)tanhf(acc11[r] + bn1);
  }
}

// ---------------- FC head: out = Hin*Wfc^T + bfc ----------------
// writes f32 slice out[:, step, :] of (B,F,C) and f16 decoder-feedback buffer
__global__ void __launch_bounds__(256)
fc_kernel(const _Float16* __restrict__ Hin,
          const _Float16* __restrict__ Wfc,   // (C, H) row-major
          const float* __restrict__ bfc,      // (C)
          float* __restrict__ out, int step,
          _Float16* __restrict__ dec) {
  const int NT = RNN_C / 16;  // 4 N-tiles
  int wave  = blockIdx.x * (blockDim.x >> 5) + (threadIdx.x >> 5);
  int tileM = wave / NT;
  int tileN = wave % NT;
  int lane  = threadIdx.x & 31;
  int l15   = lane & 15;
  int ksel  = (lane >> 4) << 3;
  int mrow  = tileM * 16 + l15;
  int ncol  = tileN * 16 + l15;

  v8f acc = {};
  const _Float16* aP = Hin + (size_t)mrow * RNN_H + ksel;
  const _Float16* bP = Wfc + (size_t)ncol * RNN_H + ksel;
  for (int k = 0; k < RNN_H; k += 32) {
    v16h a = load_frag16(aP + k);
    v16h b = load_frag16(bP + k);
    acc = __builtin_amdgcn_wmma_f32_16x16x32_f16(
        false, a, false, b, (short)0, acc, false, false);
  }

  float bn    = bfc[ncol];
  int   mbase = tileM * 16 + ((lane < 16) ? 0 : 8);
#pragma unroll
  for (int r = 0; r < 8; ++r) {
    int   m = mbase + r;
    float v = acc[r] + bn;
    out[(size_t)m * (RNN_F * RNN_C) + (size_t)step * RNN_C + ncol] = v;
    dec[(size_t)m * RNN_C + ncol] = (_Float16)v;
  }
}

// ---------------- host-side orchestration ----------------

extern "C" void kernel_launch(void* const* d_in, const int* in_sizes, int n_in,
                              void* d_out, int out_size, void* d_ws, size_t ws_size,
                              hipStream_t stream) {
  (void)in_sizes; (void)n_in; (void)out_size; (void)ws_size;
  const float* x    = (const float*)d_in[0];   // (B,T,I)
  const float* Wih0 = (const float*)d_in[1];   // (H,I)
  const float* Whh0 = (const float*)d_in[2];   // (H,H)
  const float* bih0 = (const float*)d_in[3];
  const float* bhh0 = (const float*)d_in[4];
  const float* Wih1 = (const float*)d_in[5];   // (H,H)
  const float* Whh1 = (const float*)d_in[6];   // (H,H)
  const float* bih1 = (const float*)d_in[7];
  const float* bhh1 = (const float*)d_in[8];
  const float* Wfc  = (const float*)d_in[9];   // (C,H)
  const float* bfc  = (const float*)d_in[10];
  float* out = (float*)d_out;                  // (B,F,C)

  // workspace carve-out (256B aligned)
  char*  ws  = (char*)d_ws;
  size_t off = 0;
  auto alloc = [&](size_t bytes) -> void* {
    void* p = ws + off;
    off = (off + bytes + 255) & ~(size_t)255;
    return p;
  };
  _Float16* xh    = (_Float16*)alloc((size_t)RNN_B * RNN_T * RNN_I * 2);
  _Float16* w0h   = (_Float16*)alloc((size_t)RNN_H * RNN_I * 2);
  _Float16* whh0h = (_Float16*)alloc((size_t)RNN_H * RNN_H * 2);
  _Float16* wih1h = (_Float16*)alloc((size_t)RNN_H * RNN_H * 2);
  _Float16* whh1h = (_Float16*)alloc((size_t)RNN_H * RNN_H * 2);
  _Float16* wfch  = (_Float16*)alloc((size_t)RNN_C * RNN_H * 2);
  float*    bias0 = (float*)alloc((size_t)RNN_H * 4);
  float*    bias1 = (float*)alloc((size_t)RNN_H * 4);
  _Float16* h0a   = (_Float16*)alloc((size_t)RNN_B * RNN_H * 2);
  _Float16* h0b   = (_Float16*)alloc((size_t)RNN_B * RNN_H * 2);
  _Float16* h1a   = (_Float16*)alloc((size_t)RNN_B * RNN_H * 2);
  _Float16* h1b   = (_Float16*)alloc((size_t)RNN_B * RNN_H * 2);
  _Float16* dec   = (_Float16*)alloc((size_t)RNN_B * RNN_C * 2);

  auto cvt = [&](const float* s, _Float16* d, int n) {
    f32_to_f16_kernel<<<(n + 255) / 256, 256, 0, stream>>>(s, d, n);
  };
  cvt(x,    xh,    RNN_B * RNN_T * RNN_I);
  cvt(Wih0, w0h,   RNN_H * RNN_I);
  cvt(Whh0, whh0h, RNN_H * RNN_H);
  cvt(Wih1, wih1h, RNN_H * RNN_H);
  cvt(Whh1, whh1h, RNN_H * RNN_H);
  cvt(Wfc,  wfch,  RNN_C * RNN_H);
  bias_combine_kernel<<<(RNN_H + 255) / 256, 256, 0, stream>>>(bih0, bhh0, bias0, RNN_H);
  bias_combine_kernel<<<(RNN_H + 255) / 256, 256, 0, stream>>>(bih1, bhh1, bias1, RNN_H);

  // zero initial hidden state (ws is poisoned; must re-init every call)
  zero_f16_kernel<<<(RNN_B * RNN_H + 255) / 256, 256, 0, stream>>>(h0a, RNN_B * RNN_H);
  zero_f16_kernel<<<(RNN_B * RNN_H + 255) / 256, 256, 0, stream>>>(h1a, RNN_B * RNN_H);

  // 2x2-blocked layer kernel: (B/32)*(H/32) = 8*32 = 256 waves -> 32 blocks
  const int layerBlocks = (RNN_B / 32) * (RNN_H / 32) / 8;  // 32
  const int fcBlocks    = (RNN_B / 16) * (RNN_C / 16) / 8;  // 8

  _Float16 *h0p = h0a, *h0n = h0b, *h1p = h1a, *h1n = h1b;

  // encoder: scan over T timesteps (x is (B,T,I): row stride T*I, offset t*I)
  for (int t = 0; t < RNN_T; ++t) {
    rnn_layer_kernel<<<layerBlocks, 256, 0, stream>>>(
        xh + (size_t)t * RNN_I, RNN_T * RNN_I, RNN_I,
        w0h, h0p, whh0h, bias0, h0n);
    rnn_layer_kernel<<<layerBlocks, 256, 0, stream>>>(
        h0n, RNN_H, RNN_H,
        wih1h, h1p, whh1h, bias1, h1n);
    _Float16* t0 = h0p; h0p = h0n; h0n = t0;
    _Float16* t1 = h1p; h1p = h1n; h1n = t1;
  }

  // first output from encoder's final top hidden
  fc_kernel<<<fcBlocks, 256, 0, stream>>>(h1p, wfch, bfc, out, 0, dec);

  // autoregressive decode: feed fc output (B,C) back as next input
  for (int f = 1; f < RNN_F; ++f) {
    rnn_layer_kernel<<<layerBlocks, 256, 0, stream>>>(
        dec, RNN_C, RNN_C,
        w0h, h0p, whh0h, bias0, h0n);
    rnn_layer_kernel<<<layerBlocks, 256, 0, stream>>>(
        h0n, RNN_H, RNN_H,
        wih1h, h1p, whh1h, bias1, h1n);
    _Float16* t0 = h0p; h0p = h0n; h0n = t0;
    _Float16* t1 = h1p; h1p = h1n; h1n = t1;
    fc_kernel<<<fcBlocks, 256, 0, stream>>>(h1p, wfch, bfc, out, f, dec);
  }
}
